// Prior_22119081574563
// MI455X (gfx1250) — compile-verified, matrix operand order checked by reading, not run on hardware
//
#include <hip/hip_runtime.h>

typedef __attribute__((ext_vector_type(16))) __bf16 v16bf;
typedef __attribute__((ext_vector_type(8)))  float  v8f;

__device__ __forceinline__ __bf16 f2bf(float f) { return (__bf16)f; }

// ---------------------------------------------------------------------------
// Degree / normalization
// ---------------------------------------------------------------------------
__global__ void deg_init_kernel(float* deg, int n) {
  int i = blockIdx.x * blockDim.x + threadIdx.x;
  if (i < n) deg[i] = 1.0f;  // self loop
}

__global__ void deg_accum_kernel(const int* __restrict__ dst, float* deg, int E) {
  int i = blockIdx.x * blockDim.x + threadIdx.x;
  if (i < E) atomicAdd(&deg[dst[i]], 1.0f);
}

__global__ void dinv_kernel(float* deg, int n) {
  int i = blockIdx.x * blockDim.x + threadIdx.x;
  if (i < n) deg[i] = rsqrtf(deg[i]);  // deg >= 1 always (self loop)
}

// ---------------------------------------------------------------------------
// GCN propagation: out = sum over edges of x[src] * dinv[src]*dinv[dst], plus
// self loop x[i]*dinv[i]^2. prop_self also serves as the buffer initializer.
// ---------------------------------------------------------------------------
__global__ void prop_self_kernel(const float* __restrict__ x,
                                 const float* __restrict__ dinv,
                                 float* __restrict__ out, int n) {
  size_t t = (size_t)blockIdx.x * blockDim.x + threadIdx.x;
  if (t >= (size_t)n * 128) return;
  int i = (int)(t >> 7);
  float w = dinv[i];
  out[t] = x[t] * w * w;
}

__global__ void prop_edges_kernel(const float* __restrict__ x,
                                  const int* __restrict__ src,
                                  const int* __restrict__ dst,
                                  const float* __restrict__ dinv,
                                  float* __restrict__ out, int E) {
  // one wave32 per edge; each lane handles 4 consecutive features (128 total)
  size_t t = (size_t)blockIdx.x * blockDim.x + threadIdx.x;
  if (t >= (size_t)E * 32) return;
  int e  = (int)(t >> 5);
  int c4 = (int)(t & 31) << 2;
  int s = src[e], d = dst[e];
  float w = dinv[s] * dinv[d];
  const float4 v = *(const float4*)(x + (size_t)s * 128 + c4);
  float* o = out + (size_t)d * 128 + c4;
  atomicAdd(o + 0, v.x * w);
  atomicAdd(o + 1, v.y * w);
  atomicAdd(o + 2, v.z * w);
  atomicAdd(o + 3, v.w * w);
}

// ---------------------------------------------------------------------------
// Weight pre-pack into the bf16 WMMA B-fragment lane layout (ISA 05_wmma
// 7.12.2, 16-bit K-pair packing):
//   fragment index = ((kc*CT + ct)*32 + lane)*16 + e
//   half = lane>>4 ; pair i = e>>1 ; lo = e&1
//   k   = kc*32 + ((i<4)?0:16) + (i&3)*2 + half*8 + lo
//   col = ct*16 + (lane&15)
// ---------------------------------------------------------------------------
__global__ void pack_w_kernel(const float* __restrict__ W, __bf16* __restrict__ Wp,
                              int Dout, int CT) {
  int t = blockIdx.x * blockDim.x + threadIdx.x;
  int total = 4 * CT * 32 * 16;
  if (t >= total) return;
  int e    = t & 15;
  int lane = (t >> 4) & 31;
  int ct   = (t >> 9) % CT;
  int kc   = t / (512 * CT);
  int half = lane >> 4;
  int i = e >> 1, lo = e & 1;
  int kk = ((i < 4) ? 0 : 16) + (i & 3) * 2 + half * 8 + lo;
  int k  = kc * 32 + kk;
  int col = ct * 16 + (lane & 15);
  Wp[t] = f2bf(W[(size_t)k * Dout + col]);
}

// ---------------------------------------------------------------------------
// WMMA GEMM: C[nRows x Dout] = A[nRows x 128] @ W[128 x Dout] + bias.
// Dout = CT*16 is a compile-time constant so store addressing collapses to
// one base + immediate offsets. One block (CT waves) per 16-row tile:
//   stage 1: async-copy the contiguous 16x128 fp32 tile into LDS
//            (global_load_async_to_lds_b128, ASYNCcnt)
//   stage 2: convert once to bf16 A-fragments in LDS (shared by all CT waves)
//   stage 3: each wave ct does 4 x v_wmma_f32_16x16x32_bf16 against its
//            pre-packed B fragments and writes its 16x16 tile + bias.
// ---------------------------------------------------------------------------
template <int CT>
__global__ void gemm_bias_kernel(const float* __restrict__ A,
                                 const __bf16* __restrict__ Wp,
                                 const float* __restrict__ bias,
                                 float* __restrict__ C,
                                 int nRows) {
  constexpr int Dout = CT * 16;
  __shared__ float  sAraw[16 * 128];        // 8 KB raw fp32 tile
  __shared__ __bf16 sAfrag[4 * 32 * 16];    // 4 KB bf16 fragments (kc, lane, e)

  const int tid = threadIdx.x;
  const int lane = tid & 31;
  constexpr int nThreads = CT * 32;
  const int rowBase = blockIdx.x << 4;

  // ---- stage 1: async global -> LDS copy of the 16x128 row tile ----
  unsigned rawBase = (unsigned)(uintptr_t)&sAraw[0];  // low 32 bits = LDS offset
  const char* Abytes = (const char*)A;
#pragma unroll
  for (int ofs = tid * 16; ofs < 16 * 512; ofs += nThreads * 16) {
    int r = ofs >> 9;  // 512 bytes per row
    int c = ofs & 511;
    int gr = rowBase + r;
    if (gr >= nRows) gr = nRows - 1;  // safe duplicate; stores are guarded
    unsigned long long g = (unsigned long long)(Abytes + (size_t)gr * 512 + c);
    asm volatile("global_load_async_to_lds_b128 %0, %1, off"
                 :: "v"(rawBase + (unsigned)ofs), "v"(g) : "memory");
  }
  asm volatile("s_wait_asynccnt 0x0" ::: "memory");
  __syncthreads();

  // ---- stage 2: one shared bf16 fragment conversion for all waves ----
#pragma unroll
  for (int idx = tid; idx < 2048; idx += nThreads) {
    int e    = idx & 15;
    int l    = (idx >> 4) & 31;
    int kc   = idx >> 9;
    int half = l >> 4;
    int i = e >> 1, lo = e & 1;
    int k = kc * 32 + ((i < 4) ? 0 : 16) + (i & 3) * 2 + half * 8 + lo;
    sAfrag[idx] = f2bf(sAraw[(l & 15) * 128 + k]);
  }
  __syncthreads();

  // ---- stage 3: WMMA per wave (wave ct owns columns [ct*16, ct*16+16)) ----
  const int ct = tid >> 5;
  v8f acc = {};
#pragma unroll
  for (int kc = 0; kc < 4; ++kc) {
    v16bf a = *(const v16bf*)&sAfrag[(kc * 32 + lane) << 4];
    v16bf b = *(const v16bf*)(Wp + ((size_t)((kc * CT + ct) * 32 + lane) << 4));
    acc = __builtin_amdgcn_wmma_f32_16x16x32_bf16(false, a, false, b,
                                                  (short)0, acc, false, false);
  }
  const int m = lane & 15, half = lane >> 4;
  const int col = (ct << 4) + m;   // C/D layout: N = lane&15 for both lane halves
  const float bv = bias[col];
  if (rowBase + 16 <= nRows) {
    // fast path: constant-stride stores with immediate offsets
    float* Cp = C + (size_t)(rowBase + half * 8) * Dout + col;
#pragma unroll
    for (int j = 0; j < 8; ++j) Cp[j * Dout] = acc[j] + bv;  // VGPR j -> M=j / M=j+8
  } else {
#pragma unroll
    for (int j = 0; j < 8; ++j) {
      int row = rowBase + j + half * 8;
      if (row < nRows) C[(size_t)row * Dout + col] = acc[j] + bv;
    }
  }
}

// ---------------------------------------------------------------------------
// BatchNorm (training mode) over node dimension, 128 features
// ---------------------------------------------------------------------------
__global__ void zero_stats_kernel(float* stats) { stats[threadIdx.x] = 0.0f; }

__global__ void bn_stats_kernel(const float* __restrict__ y, float* stats, int n) {
  __shared__ float s_sum[256];
  __shared__ float s_sq[256];
  const int ROWS = 512;
  int col = threadIdx.x & 127;
  int half = threadIdx.x >> 7;
  int r0 = blockIdx.x * ROWS;
  int rEnd = r0 + ROWS; if (rEnd > n) rEnd = n;
  float sum = 0.0f, sq = 0.0f;
  for (int r = r0 + half; r < rEnd; r += 2) {
    float v = y[(size_t)r * 128 + col];
    sum += v; sq += v * v;
  }
  s_sum[threadIdx.x] = sum; s_sq[threadIdx.x] = sq;
  __syncthreads();
  if (half == 0) {
    atomicAdd(&stats[col],       s_sum[col] + s_sum[col + 128]);
    atomicAdd(&stats[128 + col], s_sq[col]  + s_sq[col + 128]);
  }
}

__global__ void bn_finalize_kernel(const float* __restrict__ stats,
                                   const float* __restrict__ gamma,
                                   const float* __restrict__ beta,
                                   float* ss, float n) {
  int c = threadIdx.x;  // 128 threads
  float mean = stats[c] / n;
  float var  = stats[128 + c] / n - mean * mean;
  float istd = rsqrtf(var + 1e-5f);
  float sc = istd * gamma[c];
  ss[c] = sc;
  ss[128 + c] = beta[c] - mean * sc;
}

__global__ void bn_relu_kernel(float* y, const float* __restrict__ ss, size_t total) {
  size_t t = (size_t)blockIdx.x * blockDim.x + threadIdx.x;
  if (t >= total) return;
  int c = (int)(t & 127);
  float v = y[t] * ss[c] + ss[128 + c];
  y[t] = v > 0.0f ? v : 0.0f;
}

// ---------------------------------------------------------------------------
// Link decode: sigmoid(dot(emb[i], emb[j])) — one wave32 per label edge
// ---------------------------------------------------------------------------
__global__ void decode_kernel(const float* __restrict__ emb,
                              const int* __restrict__ li,
                              const int* __restrict__ lj,
                              float* __restrict__ r, int EL) {
  int wid = (int)(((size_t)blockIdx.x * blockDim.x + threadIdx.x) >> 5);
  int lane = threadIdx.x & 31;
  if (wid >= EL) return;
  const float* pa = emb + (size_t)li[wid] * 128;
  const float* pb = emb + (size_t)lj[wid] * 128;
  float s = 0.0f;
#pragma unroll
  for (int c = 0; c < 4; ++c) s += pa[lane + 32 * c] * pb[lane + 32 * c];
#pragma unroll
  for (int off = 16; off > 0; off >>= 1) s += __shfl_xor(s, off, 32);
  if (lane == 0) r[wid] = 1.0f / (1.0f + expf(-s));
}

// ---------------------------------------------------------------------------
// Orchestration
// ---------------------------------------------------------------------------
static inline unsigned cdiv(size_t a, size_t b) { return (unsigned)((a + b - 1) / b); }

extern "C" void kernel_launch(void* const* d_in, const int* in_sizes, int n_in,
                              void* d_out, int out_size, void* d_ws, size_t ws_size,
                              hipStream_t stream) {
  const float* x   = (const float*)d_in[0];
  const int*   ei  = (const int*)d_in[1];
  const int*   eli = (const int*)d_in[2];
  const float* W1  = (const float*)d_in[3];
  const float* b1  = (const float*)d_in[4];
  const float* g1  = (const float*)d_in[5];
  const float* be1 = (const float*)d_in[6];
  const float* W2  = (const float*)d_in[7];
  const float* b2  = (const float*)d_in[8];
  const float* g2  = (const float*)d_in[9];
  const float* be2 = (const float*)d_in[10];
  const float* W3  = (const float*)d_in[11];
  const float* b3  = (const float*)d_in[12];

  const int N  = in_sizes[0] / 128;
  const int E  = in_sizes[1] / 2;
  const int EL = in_sizes[2] / 2;
  const int* src = ei;
  const int* dst = ei + E;
  const int* li  = eli;
  const int* lj  = eli + EL;

  char* ws = (char*)d_ws;
  size_t off = 0;
  auto alloc = [&](size_t bytes) -> char* {
    char* p = ws + off;
    off = (off + bytes + 255) & ~(size_t)255;
    return p;
  };
  float*  dinv  = (float*)alloc((size_t)N * 4);
  float*  stats = (float*)alloc(256 * 4);
  float*  ss    = (float*)alloc(256 * 4);
  __bf16* Wp1   = (__bf16*)alloc(16384 * 2);   // 128x128, CT=8
  __bf16* Wp2   = (__bf16*)alloc(16384 * 2);   // 128x128, CT=8
  __bf16* Wp3   = (__bf16*)alloc(8192 * 2);    // 128x64,  CT=4
  float*  buf0  = (float*)alloc((size_t)N * 128 * 4);
  float*  buf1  = (float*)alloc((size_t)N * 128 * 4);

  float* xout = (float*)d_out;
  float* rout = xout + (size_t)N * 64;

  const unsigned T = 256;
  const size_t elems = (size_t)N * 128;

  // --- weight pre-pack (bf16 WMMA fragments) ---
  pack_w_kernel<<<cdiv(4 * 8 * 32 * 16, T), T, 0, stream>>>(W1, Wp1, 128, 8);
  pack_w_kernel<<<cdiv(4 * 8 * 32 * 16, T), T, 0, stream>>>(W2, Wp2, 128, 8);
  pack_w_kernel<<<cdiv(4 * 4 * 32 * 16, T), T, 0, stream>>>(W3, Wp3, 64, 4);

  // --- degrees / symmetric normalization ---
  deg_init_kernel<<<cdiv(N, T), T, 0, stream>>>(dinv, N);
  deg_accum_kernel<<<cdiv(E, T), T, 0, stream>>>(dst, dinv, E);
  dinv_kernel<<<cdiv(N, T), T, 0, stream>>>(dinv, N);

  // GEMM geometry: one block per 16-row tile, CT waves per block
  const unsigned rowTiles = (unsigned)((N + 15) / 16);

  // ---- layer 1 ----
  prop_self_kernel<<<cdiv(elems, T), T, 0, stream>>>(x, dinv, buf0, N);
  prop_edges_kernel<<<cdiv((size_t)E * 32, T), T, 0, stream>>>(x, src, dst, dinv, buf0, E);
  gemm_bias_kernel<8><<<rowTiles, 8 * 32, 0, stream>>>(buf0, Wp1, b1, buf1, N);
  zero_stats_kernel<<<1, 256, 0, stream>>>(stats);
  bn_stats_kernel<<<cdiv(N, 512), T, 0, stream>>>(buf1, stats, N);
  bn_finalize_kernel<<<1, 128, 0, stream>>>(stats, g1, be1, ss, (float)N);
  bn_relu_kernel<<<cdiv(elems, T), T, 0, stream>>>(buf1, ss, elems);

  // ---- layer 2 ----
  prop_self_kernel<<<cdiv(elems, T), T, 0, stream>>>(buf1, dinv, buf0, N);
  prop_edges_kernel<<<cdiv((size_t)E * 32, T), T, 0, stream>>>(buf1, src, dst, dinv, buf0, E);
  gemm_bias_kernel<8><<<rowTiles, 8 * 32, 0, stream>>>(buf0, Wp2, b2, buf1, N);
  zero_stats_kernel<<<1, 256, 0, stream>>>(stats);
  bn_stats_kernel<<<cdiv(N, 512), T, 0, stream>>>(buf1, stats, N);
  bn_finalize_kernel<<<1, 128, 0, stream>>>(stats, g2, be2, ss, (float)N);
  bn_relu_kernel<<<cdiv(elems, T), T, 0, stream>>>(buf1, ss, elems);

  // ---- layer 3: emb = prop(h2); x_out = emb @ W3 + b3; decode ----
  prop_self_kernel<<<cdiv(elems, T), T, 0, stream>>>(buf1, dinv, buf0, N);
  prop_edges_kernel<<<cdiv((size_t)E * 32, T), T, 0, stream>>>(buf1, src, dst, dinv, buf0, E);
  gemm_bias_kernel<4><<<rowTiles, 4 * 32, 0, stream>>>(buf0, Wp3, b3, xout, N);
  decode_kernel<<<cdiv((size_t)EL * 32, T), T, 0, stream>>>(buf0, li, lj, rout, EL);
}
